// MultiHeadSelfAttention2D_4148938408365
// MI455X (gfx1250) — matrix operand order repaired and checked by reading, not compiled, not run on hardware
//
#include <hip/hip_runtime.h>

// MultiHeadSelfAttention2D for MI455X (gfx1250), bf16 WMMA pipeline.
// ~206 GFLOP vs ~270MB mandatory HBM traffic -> compute bound -> use
// v_wmma_f32_16x16x32_bf16 everywhere. All fragment loads are contiguous
// 16B per lane per the CDNA5 ISA 7.12.2 layouts. K-tile staging in the
// attention QK^T kernel uses the Tensor Data Mover (tensor_load_to_lds +
// s_wait_tensorcnt), double-buffered.
//
// Workspace usage: ~225 MB (Q/K/V bf16 67MB each, S fp32 32MB; VoR aliases Q,
// P aliases S row-slack).

typedef __bf16 bf16;
typedef __attribute__((ext_vector_type(16))) __bf16 bf16x16;
typedef __attribute__((ext_vector_type(8)))  __bf16 bf16x8;
typedef __attribute__((ext_vector_type(4)))  __bf16 bf16x4;
typedef __attribute__((ext_vector_type(8)))  float  f32x8;
typedef __attribute__((ext_vector_type(4)))  float  f32x4;
typedef __attribute__((ext_vector_type(4)))  unsigned int u32x4;
typedef __attribute__((ext_vector_type(8)))  int    i32x8;
typedef __attribute__((ext_vector_type(4)))  int    i32x4;

#define DEV static __device__ __forceinline__

DEV f32x8 wmma_bf16(bf16x16 a, bf16x16 b, f32x8 c) {
  // D = A(16x32) * B(32x16) + C, f32 accumulate
  return __builtin_amdgcn_wmma_f32_16x16x32_bf16(false, a, false, b,
                                                 (short)0, c, false, false);
}

// A fragment (16x32 bf16) from row-major [rows, ld] at (row0, k0).
// lane l holds row (l&15); K = (l>>4)*8 + {0..7} and 16 + (l>>4)*8 + {0..7}.
DEV bf16x16 load_a(const bf16* p0, int ld, int row0, int k0, int lane) {
  const bf16* p = p0 + (size_t)(row0 + (lane & 15)) * ld + k0 + ((lane >> 4) << 3);
  bf16x8 lo = *(const bf16x8*)p;
  bf16x8 hi = *(const bf16x8*)(p + 16);
  bf16x16 r;
#pragma unroll
  for (int i = 0; i < 8; ++i) { r[i] = lo[i]; r[i + 8] = hi[i]; }
  return r;
}

// B fragment (32x16 bf16) where source is stored "N-major": src[n*ld + k]
// holds B[k][n]. lane l holds col (n0 + (l&15)); K = (l>>4)*8 + {0..7} and 16 + (l>>4)*8 + {0..7}.
// Caller folds the K-chunk base into p0.
DEV bf16x16 load_b(const bf16* p0, int ld, int n0, int lane) {
  const bf16* p = p0 + (size_t)(n0 + (lane & 15)) * ld + ((lane >> 4) << 3);
  bf16x8 lo = *(const bf16x8*)p;
  bf16x8 hi = *(const bf16x8*)(p + 16);
  bf16x16 r;
#pragma unroll
  for (int i = 0; i < 8; ++i) { r[i] = lo[i]; r[i + 8] = hi[i]; }
  return r;
}

// ------------------------------------------------------------------- TDM
// 2D tile DMA (global -> LDS) per CDNA5 ISA.
// Elements are 2 bytes. tile_d0 elems per row, tile_d1 rows, global row
// stride stride_e elems. LDS dest padded pad 4 DWORDs every 16 DWORDs
// (i.e. rows of 64B data + 16B pad = 40-element pitch).
DEV void tdm_load_tile_2d(unsigned lds_byte_off, const bf16* g,
                          unsigned tile_d0, unsigned tile_d1,
                          unsigned stride_e) {
  unsigned long long ga = (unsigned long long)(size_t)g;
  const unsigned td1 = 1u << 20;            // generous tensor_dim1 (no OOB)
  u32x4 g0;
  g0[0] = 1u;                               // count=1, user descriptor
  g0[1] = lds_byte_off;                     // lds_addr (bytes)
  g0[2] = (unsigned)ga;                     // global_addr[31:0]
  g0[3] = ((unsigned)(ga >> 32) & 0x01FFFFFFu) | 0x80000000u; // addr hi, type=2
  i32x8 g1;
  g1[0] = (int)((1u << 16)                  // data_size = 2 bytes
                | (1u << 20)                // pad_enable
                | (3u << 22)                // pad_interval: 16 DWORDs
                | (3u << 25));              // pad_amount: 4 DWORDs (16B)
  g1[1] = (int)((stride_e & 0xFFFFu) << 16);            // tensor_dim0 lo16
  g1[2] = (int)(((stride_e >> 16) & 0xFFFFu)            // tensor_dim0 hi16
                | ((td1 & 0xFFFFu) << 16));             // tensor_dim1 lo16
  g1[3] = (int)(((td1 >> 16) & 0xFFFFu)                 // tensor_dim1 hi16
                | (tile_d0 << 16));                     // tile_dim0
  g1[4] = (int)(tile_d1 & 0xFFFFu);                     // tile_dim1 (dim2=0)
  g1[5] = (int)stride_e;                                // tensor_dim0_stride lo
  g1[6] = 0;                                            // stride hi, dim1_stride
  g1[7] = 0;
  i32x4 z4 = {};
#if defined(__clang_major__) && (__clang_major__ >= 23)
  i32x8 z8 = {};
  __builtin_amdgcn_tensor_load_to_lds(g0, g1, z4, z4, z8, 0);
#else
  __builtin_amdgcn_tensor_load_to_lds(g0, g1, z4, z4, 0);
#endif
}

DEV unsigned lds_off(const void* p) {
  // flat LDS aperture: addr[31:0] is the LDS byte offset
  return (unsigned)(size_t)p;
}

// ---------------------------------------------------------------- weights
__global__ __launch_bounds__(256)
void k_weights(const float* __restrict__ Wq, const float* __restrict__ Wk,
               const float* __restrict__ Wv, const float* __restrict__ Wp,
               bf16* __restrict__ Wqkv, bf16* __restrict__ Wpb) {
  int i = blockIdx.x * 256 + threadIdx.x;
  if (i < 65536)       Wqkv[i] = (bf16)Wq[i];
  else if (i < 131072) Wqkv[i] = (bf16)Wk[i - 65536];
  else if (i < 196608) Wqkv[i] = (bf16)Wv[i - 131072];
  if (i < 65536)       Wpb[i] = (bf16)Wp[i];
}

// ------------------------------------------------------------- projection
// QKV[b] = Wqkv(768x256) @ x[b](256 x 65536), col tile = one t (64 f cols).
// Q[m][t][h*64+f], K same, V[m][t][h'*64+f]  (m = n*2 + b), bf16.
__global__ __launch_bounds__(256)
void k_proj(const float* __restrict__ x, const bf16* __restrict__ Wqkv,
            const float* __restrict__ bq, const float* __restrict__ bk,
            const float* __restrict__ bv,
            bf16* __restrict__ Q, bf16* __restrict__ K, bf16* __restrict__ V) {
  __shared__ bf16 xT[64][264];   // [f][c], padded pitch
  const int t = blockIdx.x, b = blockIdx.y;
  const int tid = threadIdx.x;
  {
    const float* xp = x + ((size_t)(b * 256 + tid) * 1024 + t) * 64;
#pragma unroll
    for (int f4 = 0; f4 < 64; f4 += 4) {
      f32x4 v = *(const f32x4*)(xp + f4);
#pragma unroll
      for (int i = 0; i < 4; ++i) xT[f4 + i][tid] = (bf16)v[i];
    }
  }
  __syncthreads();
  const int w = tid >> 5, lane = tid & 31;
  for (int g = 0; g < 3; ++g) {           // 3 row-groups of 32 rows per wave
    f32x8 acc[2][4] = {};
    for (int kc = 0; kc < 8; ++kc) {
      const int k0 = kc * 32;
      bf16x16 a0 = load_a(Wqkv, 256, w * 96 + g * 32, k0, lane);
      bf16x16 a1 = load_a(Wqkv, 256, w * 96 + g * 32 + 16, k0, lane);
#pragma unroll
      for (int j = 0; j < 4; ++j) {
        bf16x16 bb = load_b(&xT[0][0] + k0, 264, j * 16, lane);
        acc[0][j] = wmma_bf16(a0, bb, acc[0][j]);
        acc[1][j] = wmma_bf16(a1, bb, acc[1][j]);
      }
    }
#pragma unroll
    for (int i = 0; i < 2; ++i) {
      const int tile = w * 96 + g * 32 + i * 16;
      const int which = tile >> 8;        // 0:Q 1:K 2:V (tiles never straddle)
      const int rbase = tile & 255;
      bf16* dst; const float* bias;
      if (which == 0)      { dst = Q; bias = bq; }
      else if (which == 1) { dst = K; bias = bk; }
      else                 { dst = V; bias = bv; }
#pragma unroll
      for (int j = 0; j < 4; ++j) {
        const int f = j * 16 + (lane & 15);
#pragma unroll
        for (int r = 0; r < 8; ++r) {
          const int rr = rbase + r + 8 * (lane >> 4);
          const int n = rr >> 6, h = rr & 63;
          const int mm = n * 2 + b;
          dst[((size_t)(mm * 1024 + t)) * 4096 + h * 64 + f] =
              (bf16)(acc[i][j][r] + bias[rr]);
        }
      }
    }
  }
}

// ------------------------------------------------------ S = scale * Q K^T
// grid (sb=4, tb=8, m=8): 256 s cols x 128 t rows per WG.
// K tiles [256 s][32 e] DMA'd to LDS by the TDM, ping-pong buffered; pad
// descriptor reproduces the 40-element LDS pitch.
__global__ __launch_bounds__(256)
void k_qk(const bf16* __restrict__ Q, const bf16* __restrict__ Kmat,
          float* __restrict__ S) {
  __shared__ bf16 KtF[2][256 * 40];       // [s_local*40 + e_local]
  const int sb = blockIdx.x, tb = blockIdx.y, m = blockIdx.z;
  const int tid = threadIdx.x, w = tid >> 5, lane = tid & 31;
  const bf16* Kbase = Kmat + ((size_t)(m * 1024 + sb * 256)) * 4096;
  const bf16* Qrow  = Q + ((size_t)(m * 1024 + tb * 128 + w * 16)) * 4096;
  if (tid < 32) {
    tdm_load_tile_2d(lds_off(&KtF[0][0]), Kbase, 32, 256, 4096);
    __builtin_amdgcn_s_wait_tensorcnt(0);
  }
  __syncthreads();
  f32x8 acc[16] = {};
  for (int kc = 0; kc < 128; ++kc) {
    const int cur = kc & 1;
    if (tid < 32 && kc + 1 < 128)
      tdm_load_tile_2d(lds_off(&KtF[cur ^ 1][0]), Kbase + (kc + 1) * 32,
                       32, 256, 4096);
    if (kc + 2 < 128)
      __builtin_prefetch(Qrow + (size_t)(lane & 15) * 4096 + (kc + 2) * 32, 0, 3);
    bf16x16 a = load_a(Qrow, 4096, 0, kc * 32, lane);
#pragma unroll
    for (int j = 0; j < 16; ++j) {
      bf16x16 bb = load_b(&KtF[cur][0], 40, j * 16, lane);
      acc[j] = wmma_bf16(a, bb, acc[j]);
    }
    if (tid < 32) __builtin_amdgcn_s_wait_tensorcnt(0);
    __syncthreads();
  }
  const float scale = 0.015625f;          // 1/sqrt(4096)
  const int half = lane >> 4;
#pragma unroll
  for (int j = 0; j < 16; ++j) {
    const int s_g = sb * 256 + j * 16 + (lane & 15);
#pragma unroll
    for (int r = 0; r < 8; ++r) {
      const int t_g = tb * 128 + w * 16 + r + 8 * half;
      S[((size_t)(m * 1024 + t_g)) * 1024 + s_g] = acc[j][r] * scale;
    }
  }
}

// ----------------------------------------------------------------- softmax
// one wave32 per row of S[m*1024+t][1024]; P (bf16) written in-place into the
// first 2KB of each 4KB S row.
__global__ __launch_bounds__(256)
void k_softmax(float* __restrict__ S) {
  const int w = threadIdx.x >> 5, lane = threadIdx.x & 31;
  float* Srow = S + ((size_t)blockIdx.x * 8 + w) * 1024;
  f32x4 v[8];
  float mx = -3.4e38f;
#pragma unroll
  for (int i = 0; i < 8; ++i) {
    v[i] = *(const f32x4*)(Srow + i * 128 + lane * 4);
#pragma unroll
    for (int k = 0; k < 4; ++k) mx = fmaxf(mx, v[i][k]);
  }
#pragma unroll
  for (int off = 16; off > 0; off >>= 1) mx = fmaxf(mx, __shfl_xor(mx, off));
  float sum = 0.f;
#pragma unroll
  for (int i = 0; i < 8; ++i)
#pragma unroll
    for (int k = 0; k < 4; ++k) { float e = __expf(v[i][k] - mx); v[i][k] = e; sum += e; }
#pragma unroll
  for (int off = 16; off > 0; off >>= 1) sum += __shfl_xor(sum, off);
  const float inv = 1.f / sum;
  bf16* Prow = (bf16*)Srow;               // in-place; reads already in regs
#pragma unroll
  for (int i = 0; i < 8; ++i) {
    bf16x4 pv;
#pragma unroll
    for (int k = 0; k < 4; ++k) pv[k] = (bf16)(v[i][k] * inv);
    *(bf16x4*)(Prow + i * 128 + lane * 4) = pv;
  }
}

// ------------------------------------------------------------------- P @ V
// grid (db=64, tb=8, m=8). Epilogue folds the reference's transpose/reshape:
// VoR[b][c=n*64+hp][tt][f], tt = (d&63)*16 + t>>6, f = t&63, hp = d>>6.
__global__ __launch_bounds__(256)
void k_pv(const bf16* __restrict__ P, const bf16* __restrict__ V,
          bf16* __restrict__ VoR) {
  __shared__ bf16 Vt[2][64][40];          // [d_local][s_local], pitch 40
  const int db = blockIdx.x, tb = blockIdx.y, m = blockIdx.z;
  const int tid = threadIdx.x, w = tid >> 5, lane = tid & 31;
  const int sl = tid >> 3, d8 = (tid & 7) * 8;
  const bf16* Vbase = V + ((size_t)(m * 1024) + sl) * 4096 + db * 64 + d8;
  const bf16* Prow  = P + ((size_t)(m * 1024 + tb * 128 + w * 16)) * 2048;
  {
    bf16x8 vv = *(const bf16x8*)Vbase;
#pragma unroll
    for (int i = 0; i < 8; ++i) Vt[0][d8 + i][sl] = vv[i];
  }
  __syncthreads();
  f32x8 acc[4] = {};
  for (int kc = 0; kc < 32; ++kc) {
    const int cur = kc & 1;
    if (kc + 1 < 32) {
      bf16x8 vv = *(const bf16x8*)(Vbase + (size_t)(kc + 1) * 32 * 4096);
#pragma unroll
      for (int i = 0; i < 8; ++i) Vt[cur ^ 1][d8 + i][sl] = vv[i];
    }
    bf16x16 a = load_a(Prow, 2048, 0, kc * 32, lane);
#pragma unroll
    for (int j = 0; j < 4; ++j) {
      bf16x16 bb = load_b(&Vt[cur][0][0], 40, j * 16, lane);
      acc[j] = wmma_bf16(a, bb, acc[j]);
    }
    __syncthreads();
  }
  const int b = m & 1, n = m >> 1, hp = db;
  const int half = lane >> 4;
  const int tbase = tb * 128 + w * 16;
  const int fbase = (tbase & 63) + 8 * half;   // f = fbase + r (no carry)
  const int thi = tbase >> 6;
#pragma unroll
  for (int j = 0; j < 4; ++j) {
    const int delta = j * 16 + (lane & 15);
    const int tt = delta * 16 + thi;
    bf16x8 o;
#pragma unroll
    for (int r = 0; r < 8; ++r) o[r] = (bf16)acc[j][r];
    *(bf16x8*)(VoR + ((size_t)((b * 256 + n * 64 + hp) * 1024 + tt)) * 64 + fbase) = o;
  }
}

// ------------------------------------------------------- output projection
// out[b][o][tt][f] = Wp[o][:] . VoR[b][:][tt][f] + bp[o] + x[b][o][tt][f]
__global__ __launch_bounds__(256)
void k_out(const bf16* __restrict__ VoR, const bf16* __restrict__ Wpb,
           const float* __restrict__ bp, const float* __restrict__ x,
           float* __restrict__ out) {
  __shared__ bf16 Bt[64][264];            // [f][c]
  const int tt = blockIdx.x, b = blockIdx.y;
  const int tid = threadIdx.x, w = tid >> 5, lane = tid & 31;
  {
    const bf16* vp = VoR + ((size_t)(b * 256 + tid) * 1024 + tt) * 64;
#pragma unroll
    for (int i = 0; i < 8; ++i) {
      bf16x8 vv = *(const bf16x8*)(vp + i * 8);
#pragma unroll
      for (int k = 0; k < 8; ++k) Bt[i * 8 + k][tid] = vv[k];
    }
  }
  __syncthreads();
  f32x8 acc[2][4] = {};
  for (int kc = 0; kc < 8; ++kc) {
    const int k0 = kc * 32;
    bf16x16 a0 = load_a(Wpb, 256, w * 32, k0, lane);
    bf16x16 a1 = load_a(Wpb, 256, w * 32 + 16, k0, lane);
#pragma unroll
    for (int j = 0; j < 4; ++j) {
      bf16x16 bb = load_b(&Bt[0][0] + k0, 264, j * 16, lane);
      acc[0][j] = wmma_bf16(a0, bb, acc[0][j]);
      acc[1][j] = wmma_bf16(a1, bb, acc[1][j]);
    }
  }
  const int half = lane >> 4;
#pragma unroll
  for (int i = 0; i < 2; ++i)
#pragma unroll
    for (int j = 0; j < 4; ++j) {
      const int f = j * 16 + (lane & 15);
#pragma unroll
      for (int r = 0; r < 8; ++r) {
        const int o = w * 32 + i * 16 + r + 8 * half;
        const size_t idx = ((size_t)(b * 256 + o) * 1024 + tt) * 64 + f;
        out[idx] = acc[i][j][r] + bp[o] + x[idx];
      }
    }
}

// ------------------------------------------------------------------ launch
extern "C" void kernel_launch(void* const* d_in, const int* in_sizes, int n_in,
                              void* d_out, int out_size, void* d_ws, size_t ws_size,
                              hipStream_t stream) {
  (void)in_sizes; (void)n_in; (void)out_size; (void)ws_size;
  const float* x  = (const float*)d_in[0];
  const float* Wq = (const float*)d_in[1];
  const float* bq = (const float*)d_in[2];
  const float* Wk = (const float*)d_in[3];
  const float* bk = (const float*)d_in[4];
  const float* Wv = (const float*)d_in[5];
  const float* bv = (const float*)d_in[6];
  const float* Wp = (const float*)d_in[7];
  const float* bp = (const float*)d_in[8];
  float* out = (float*)d_out;

  char* ws = (char*)d_ws;
  bf16*  Wqkv = (bf16*)ws;                       // 768*256*2      = 393216
  bf16*  Wpb  = (bf16*)(ws + 393216);            // 256*256*2      = 131072
  bf16*  Q    = (bf16*)(ws + 524288);            // 8*1024*4096*2  = 67108864
  bf16*  Kb   = (bf16*)(ws + 67633152);          // 67108864
  bf16*  Vb   = (bf16*)(ws + 134742016);         // 67108864
  float* S    = (float*)(ws + 201850880);        // 8*1024*1024*4  = 33554432
  bf16*  P    = (bf16*)S;                        // alias, row pitch 2048 bf16
  bf16*  VoR  = Q;                               // alias (Q dead after k_qk)
  // total workspace required: 235,405,312 bytes

  k_weights<<<768, 256, 0, stream>>>(Wq, Wk, Wv, Wp, Wqkv, Wpb);
  k_proj<<<dim3(1024, 2), 256, 0, stream>>>(x, Wqkv, bq, bk, bv, Q, Kb, Vb);
  k_qk<<<dim3(4, 8, 8), 256, 0, stream>>>(Q, Kb, S);
  k_softmax<<<1024, 256, 0, stream>>>(S);
  k_pv<<<dim3(64, 8, 8), 256, 0, stream>>>(P, Vb, VoR);
  k_out<<<dim3(1024, 2), 256, 0, stream>>>(VoR, Wpb, bp, x, out);
}
